// CausalAttention_27608049778749
// MI455X (gfx1250) — matrix operand order repaired and checked by reading, not compiled, not run on hardware
//
#include <hip/hip_runtime.h>

// ---------------------------------------------------------------------------
// CDNA5 (gfx1250) causal attention: bf16 WMMA pipeline
//   stage 1: cast x, Wq, Wk, Wv  f32 -> bf16 (workspace)
//   stage 2: QKV projection GEMM (wmma_f32_16x16x32_bf16), V stored transposed
//   stage 3: fused scores+softmax+PV per 16-query tile, S/P staged in LDS,
//            Q tile staged via global_load_async_to_lds_b128 (ASYNCcnt),
//            register-blocked, branch-free hot loops (8 WMMAs per A-read)
// ---------------------------------------------------------------------------

typedef __bf16 v16bf __attribute__((ext_vector_type(16)));
typedef __bf16 v8bf  __attribute__((ext_vector_type(8)));
typedef float  v8f   __attribute__((ext_vector_type(8)));

static constexpr int BATCH = 4;
static constexpr int SEQ   = 2048;
static constexpr int DIM   = 1024;   // d_in == d_out

// LDS pitches (elements), padded to dodge bank conflicts
static constexpr int SPITCH = SEQ + 8;   // f32 scores row
static constexpr int PPITCH = SEQ + 8;   // bf16 probs row
static constexpr int QPITCH = DIM + 8;   // bf16 staged Q row
static constexpr int LDS_BYTES = 16 * SPITCH * 4 + 16 * PPITCH * 2 + 16 * QPITCH * 2;

__device__ __forceinline__ unsigned short f2bf(float f) {
    unsigned int u = __float_as_uint(f);
    unsigned int r = u + 0x7FFFu + ((u >> 16) & 1u);   // round-to-nearest-even
    return (unsigned short)(r >> 16);
}

__device__ __forceinline__ v16bf concat8(v8bf lo, v8bf hi) {
    return __builtin_shufflevector(lo, hi, 0, 1, 2, 3, 4, 5, 6, 7,
                                           8, 9, 10, 11, 12, 13, 14, 15);
}

// A-matrix 16x32 bf16: lanes 0-15 row M=lane, K=k0..k0+15 contiguous;
// lanes 16-31 row M=lane-16, K=k0+16..k0+31.
// `aBase` is pre-offset by row*pitch + (lane&16 ? 16 : 0).
__device__ __forceinline__ v16bf fragA(const unsigned short* aBase, int k0) {
    return *reinterpret_cast<const v16bf*>(aBase + k0);
}

// B-matrix 32x16 bf16: lanes 0-15 hold column N=lane, K=k0..k0+15 contiguous;
// lanes 16-31 hold column N=lane-16, K=k0+16..k0+31.
// `bBase` is pre-offset by N*pitch + (lane&16 ? 16 : 0).
__device__ __forceinline__ v16bf fragB(const unsigned short* bBase, int k0) {
    return *reinterpret_cast<const v16bf*>(bBase + k0);
}

__device__ __forceinline__ v8f wmma_bf16(v16bf a, v16bf b, v8f c) {
    return __builtin_amdgcn_wmma_f32_16x16x32_bf16(false, a, false, b,
                                                   (short)0, c, false, false);
}

// ---------------------------------------------------------------------------
__global__ void cast_f32_bf16_kernel(const float* __restrict__ in,
                                     unsigned short* __restrict__ out, int n) {
    int i = blockIdx.x * blockDim.x + threadIdx.x;
    const int stride = gridDim.x * blockDim.x;
    for (; i < n; i += stride) out[i] = f2bf(in[i]);
}

// ---------------------------------------------------------------------------
// Each wave computes one 16x16 tile of Q, K and V simultaneously (A fragment
// of x reused 3x per K-step). Q,K stored bf16 [b,n,d]; V stored bf16 [b,d,n].
__global__ void qkv_gemm_kernel(const unsigned short* __restrict__ xb,
                                const unsigned short* __restrict__ wq,
                                const unsigned short* __restrict__ wk,
                                const unsigned short* __restrict__ wv,
                                unsigned short* __restrict__ qb,
                                unsigned short* __restrict__ kb,
                                unsigned short* __restrict__ vtb) {
    const int lane = threadIdx.x & 31;
    const int wave = threadIdx.x >> 5;
    const int gw   = blockIdx.x * 8 + wave;     // 32768 wave-tiles total
    const int b    = gw >> 13;                  // 8192 tiles per batch
    const int rem  = gw & 8191;
    const int rt   = rem >> 6;                  // 0..127 row tile (16 rows)
    const int ct   = rem & 63;                  // 0..63  col tile (16 cols)

    const int rowA  = lane & 15;
    const int koffA = (lane & 16) ? 16 : 0;
    const int koffB = (lane & 16) ? 16 : 0;

    // hoisted per-lane base pointers -> inner loads are base + imm offset
    const unsigned short* pA =
        xb + ((size_t)b * SEQ + rt * 16 + rowA) * DIM + koffA;
    const unsigned short* pBq = wq + (size_t)(ct * 16 + rowA) * DIM + koffB;
    const unsigned short* pBk = wk + (size_t)(ct * 16 + rowA) * DIM + koffB;
    const unsigned short* pBv = wv + (size_t)(ct * 16 + rowA) * DIM + koffB;

    v8f cq = {}, ck = {}, cv = {};
#pragma unroll 4
    for (int k0 = 0; k0 < DIM; k0 += 32) {
        v16bf a  = fragA(pA, k0);
        v16bf bq = fragB(pBq, k0);
        v16bf bk = fragB(pBk, k0);
        v16bf bv = fragB(pBv, k0);
        cq = wmma_bf16(a, bq, cq);
        ck = wmma_bf16(a, bk, ck);
        cv = wmma_bf16(a, bv, cv);
    }

    // C layout: VGPR r, lanes 0-15 -> M=r, N=lane; lanes 16-31 -> M=8+r.
    const int nc = lane & 15;
    const int mb = (lane & 16) ? 8 : 0;

    const size_t qk_base = ((size_t)b * SEQ + rt * 16) * DIM + ct * 16 + nc;
#pragma unroll
    for (int r = 0; r < 8; ++r) {
        qb[qk_base + (size_t)(mb + r) * DIM] = f2bf(cq[r]);
        kb[qk_base + (size_t)(mb + r) * DIM] = f2bf(ck[r]);
    }
    // Vt[(o)*SEQ + n]: for fixed lane, the 8 values are contiguous -> b128.
    union { unsigned short s[8]; uint4 v; } pk;
#pragma unroll
    for (int r = 0; r < 8; ++r) pk.s[r] = f2bf(cv[r]);
    const size_t vt_base = ((size_t)b * DIM + ct * 16 + nc) * SEQ + rt * 16 + mb;
    *reinterpret_cast<uint4*>(&vtb[vt_base]) = pk.v;
}

// ---------------------------------------------------------------------------
__device__ __forceinline__ void store_s_tile(float* S, v8f acc, int j, int i0,
                                             int nc, int mb) {
    const int col = j * 16 + nc;
#pragma unroll
    for (int r = 0; r < 8; ++r) {
        const int m = mb + r;
        float v = acc[r] * 0.03125f;            // 1/sqrt(1024)
        if (col > i0 + m) v = -3.0e38f;         // causal mask
        S[m * SPITCH + col] = v;
    }
}

// One workgroup (8 waves) per (batch, 16-query tile).
__global__ void attn_kernel(const unsigned short* __restrict__ qb,
                            const unsigned short* __restrict__ kb,
                            const unsigned short* __restrict__ vtb,
                            float* __restrict__ out) {
    extern __shared__ char smem[];
    float*          S  = reinterpret_cast<float*>(smem);                       // 16 x SPITCH f32
    unsigned short* P  = reinterpret_cast<unsigned short*>(smem + 16 * SPITCH * 4);
    unsigned short* Qs = reinterpret_cast<unsigned short*>(smem + 16 * SPITCH * 4
                                                                + 16 * PPITCH * 2);

    const int qt = blockIdx.x;      // 0..127
    const int b  = blockIdx.y;      // 0..3
    const int i0 = qt * 16;
    const int tid  = threadIdx.x;
    const int lane = tid & 31;
    const int wave = tid >> 5;

    // ---- stage Q tile (16 x 1024 bf16) into LDS via async DMA ----
    {
        const unsigned short* qsrc = qb + ((size_t)b * SEQ + i0) * DIM;
        const int row = tid >> 4;
        for (int c = (tid & 15) * 8; c < DIM; c += 16 * 8) {
            unsigned lds_off = (unsigned)(unsigned long long)(Qs + row * QPITCH + c);
            unsigned long long ga =
                (unsigned long long)(qsrc + (size_t)row * DIM + c);
            asm volatile("global_load_async_to_lds_b128 %0, %1, off"
                         :: "v"(lds_off), "v"(ga) : "memory");
        }
        asm volatile("s_wait_asynccnt 0x0" ::: "memory");
    }
    __syncthreads();

    const int ncols = (qt + 1) * 16;            // valid (causal) key columns
    const int nc = lane & 15;
    const int mb = (lane & 16) ? 8 : 0;

    // ---- S = (Q Kt) / 32, masked, into LDS ----
    // Wave owns key tiles j = wave + 8*t (two passes of up to 8 tiles).
    // Guards are k-invariant -> unswitched: branch-free 8-acc fast path,
    // per-tile remainder path for the boundary pass.
    {
        const unsigned short* aBase = Qs + (lane & 15) * QPITCH + ((lane & 16) ? 16 : 0);
        const unsigned short* bAll =
            kb + (size_t)b * SEQ * DIM + (size_t)(lane & 15) * DIM + ((lane & 16) ? 16 : 0);
        for (int half = 0; half < 2; ++half) {
            const int jbase = wave + 64 * half;         // wave-uniform
            if (jbase > qt) break;
            const int nt = (qt - jbase) / 8 + 1;        // valid tiles (1..8+)
            const unsigned short* bBase = bAll + (size_t)jbase * 16 * DIM;
            if (nt >= 8) {
                // fast path: 1 LDS A-read feeds 8 WMMAs, zero branches
                v8f acc[8] = {};
                for (int k0 = 0; k0 < DIM; k0 += 32) {
                    v16bf a = fragA(aBase, k0);
                    v16bf bv[8];
#pragma unroll
                    for (int t = 0; t < 8; ++t)
                        bv[t] = fragB(bBase + (size_t)(t * 128) * DIM, k0);
#pragma unroll
                    for (int t = 0; t < 8; ++t)
                        acc[t] = wmma_bf16(a, bv[t], acc[t]);
                }
#pragma unroll
                for (int t = 0; t < 8; ++t)
                    store_s_tile(S, acc[t], jbase + 8 * t, i0, nc, mb);
            } else {
                // remainder: at most 7 tiles, one accumulator each
                for (int t = 0; t < nt; ++t) {
                    const unsigned short* bt = bBase + (size_t)(t * 128) * DIM;
                    v8f acc = {};
#pragma unroll 4
                    for (int k0 = 0; k0 < DIM; k0 += 32)
                        acc = wmma_bf16(fragA(aBase, k0), fragB(bt, k0), acc);
                    store_s_tile(S, acc, jbase + 8 * t, i0, nc, mb);
                }
            }
        }
    }
    __syncthreads();

    // ---- softmax: row = tid>>4 handled by 16 lanes (a wave half) ----
    {
        const int row = tid >> 4, sub = tid & 15;
        float* srow = S + row * SPITCH;
        float m = -3.0e38f;
        for (int c = sub; c < ncols; c += 16) m = fmaxf(m, srow[c]);
        for (int o = 8; o; o >>= 1) m = fmaxf(m, __shfl_xor(m, o, 32));
        float s = 0.0f;
        for (int c = sub; c < ncols; c += 16) {
            float e = __expf(srow[c] - m);
            srow[c] = e;                        // cache exp for the store pass
            s += e;
        }
        for (int o = 8; o; o >>= 1) s += __shfl_xor(s, o, 32);
        const float inv = 1.0f / s;
        unsigned short* prow = P + row * PPITCH;
        for (int c = sub; c < ncols; c += 16) prow[c] = f2bf(srow[c] * inv);
    }
    // zero-pad P to a K=32 boundary so the PV loop needs no tail handling
    const int npad = (ncols + 31) & ~31;
    if (npad > ncols) {
        const int row = tid >> 4, sub = tid & 15;
        P[row * PPITCH + ncols + sub] = 0;
    }
    __syncthreads();

    // ---- O = P * V, register-blocked: wave owns ot = wave + 8*t ----
    {
        const unsigned short* aBase = P + (lane & 15) * PPITCH + ((lane & 16) ? 16 : 0);
        // B rows are Vt[o, :]; o = (wave + 8t)*16 + N = wave*16 + N + 128*t
        const unsigned short* vBase =
            vtb + (size_t)b * DIM * SEQ
                + (size_t)(wave * 16 + (lane & 15)) * SEQ + ((lane & 16) ? 16 : 0);
        v8f acc[8] = {};
        for (int k0 = 0; k0 < npad; k0 += 32) {
            v16bf a = fragA(aBase, k0);
            v16bf bv[8];
#pragma unroll
            for (int t = 0; t < 8; ++t)
                bv[t] = fragB(vBase + (size_t)(128 * t) * SEQ, k0);
#pragma unroll
            for (int t = 0; t < 8; ++t)
                acc[t] = wmma_bf16(a, bv[t], acc[t]);
        }
        float* obase = out + ((size_t)b * SEQ + i0) * DIM;
#pragma unroll
        for (int t = 0; t < 8; ++t) {
            const int ocol = (wave + 8 * t) * 16 + nc;
#pragma unroll
            for (int r = 0; r < 8; ++r)
                obase[(size_t)(mb + r) * DIM + ocol] = acc[t][r];
        }
    }
}

// ---------------------------------------------------------------------------
extern "C" void kernel_launch(void* const* d_in, const int* in_sizes, int n_in,
                              void* d_out, int out_size, void* d_ws, size_t ws_size,
                              hipStream_t stream) {
    const float* x  = (const float*)d_in[0];   // [4,2048,1024]
    const float* Wq = (const float*)d_in[1];   // [1024,1024]
    const float* Wk = (const float*)d_in[2];
    const float* Wv = (const float*)d_in[3];
    float* out = (float*)d_out;                // [4,2048,1024] f32

    const size_t NX = (size_t)BATCH * SEQ * DIM;   // 8388608
    const size_t NW = (size_t)DIM * DIM;           // 1048576

    unsigned short* xb  = (unsigned short*)d_ws;
    unsigned short* wqb = xb  + NX;
    unsigned short* wkb = wqb + NW;
    unsigned short* wvb = wkb + NW;
    unsigned short* qb  = wvb + NW;
    unsigned short* kb  = qb  + NX;
    unsigned short* vtb = kb  + NX;

    cast_f32_bf16_kernel<<<4096, 256, 0, stream>>>(x,  xb,  (int)NX);
    cast_f32_bf16_kernel<<<512,  256, 0, stream>>>(Wq, wqb, (int)NW);
    cast_f32_bf16_kernel<<<512,  256, 0, stream>>>(Wk, wkb, (int)NW);
    cast_f32_bf16_kernel<<<512,  256, 0, stream>>>(Wv, wvb, (int)NW);

    qkv_gemm_kernel<<<4096, 256, 0, stream>>>(xb, wqb, wkb, wvb, qb, kb, vtb);

    (void)hipFuncSetAttribute(reinterpret_cast<const void*>(attn_kernel),
                              hipFuncAttributeMaxDynamicSharedMemorySize,
                              LDS_BYTES);
    attn_kernel<<<dim3(128, BATCH), 256, LDS_BYTES, stream>>>(qb, kb, vtb, out);
}
